// Encoder_10952166605134
// MI455X (gfx1250) — matrix operand order repaired and checked by reading, not compiled, not run on hardware
//
#include <hip/hip_runtime.h>

// ---------------------------------------------------------------------------
// Hyperbolic GCN encoder for MI455X (gfx1250, wave32).
// Dense matvecs run on v_wmma_f32_16x16x32_bf16 (values live in (-1,1) after
// tanh, so bf16 inputs + f32 accumulate is the precision/throughput sweet
// spot). K=256 for both layers -> fully unrolled, software-pipelined K-loop
// (prefetch next-step fragments before current WMMAs), b128 fragment loads,
// 16x64 output strip per wave (4 independent WMMAs per K-step).
// Everything else (row norms, Mobius algebra, edge scatter) is fp32 and
// bandwidth/atomic bound; the ~130MB working set fits in the 192MB L2.
// ---------------------------------------------------------------------------

typedef __attribute__((ext_vector_type(16))) __bf16        v16bf;
typedef __attribute__((ext_vector_type(8)))  float         v8f;
typedef __attribute__((ext_vector_type(8)))  unsigned int  v8u;

#define MAXNORM (1.0f - 4e-3f)
#define TINY    1e-15f

__device__ __forceinline__ unsigned short f2bf(float f) {
    unsigned int u = __float_as_uint(f);
    unsigned int r = (u + 0x7FFFu + ((u >> 16) & 1u)) >> 16;   // round-to-nearest-even
    return (unsigned short)r;
}

__device__ __forceinline__ float artanh_c(float u) {
    u = fminf(fmaxf(u, -1.0f + 1e-7f), 1.0f - 1e-7f);
    return 0.5f * logf((1.0f + u) / (1.0f - u));
}

__device__ __forceinline__ float block_sum(float v, float* sm) {
    const int t = threadIdx.x;
    sm[t] = v;
    __syncthreads();
    for (int s = blockDim.x >> 1; s > 0; s >>= 1) {
        if (t < s) sm[t] += sm[t + s];
        __syncthreads();
    }
    float r = sm[0];
    __syncthreads();
    return r;
}

// ---------------------------------------------------------------------------
// small utility kernels
// ---------------------------------------------------------------------------
__global__ void fill1_kernel(float* p, int n) {
    int i = blockIdx.x * blockDim.x + threadIdx.x;
    if (i < n) p[i] = 1.0f;                    // self-loop contributes 1 to degree
}

__global__ void deg_edges_kernel(const int* __restrict__ col, float* __restrict__ deg, int E) {
    int i = blockIdx.x * blockDim.x + threadIdx.x;
    if (i < E) atomicAdd(&deg[col[i]], 1.0f);
}

__global__ void rsqrt_kernel(float* p, int n) {
    int i = blockIdx.x * blockDim.x + threadIdx.x;
    if (i < n) p[i] = rsqrtf(p[i]);            // deg >= 1 always (self loops)
}

__global__ void to_bf16_kernel(const float* __restrict__ w, unsigned short* __restrict__ o, int n) {
    int i = blockIdx.x * blockDim.x + threadIdx.x;
    if (i < n) o[i] = f2bf(w[i]);
}

// ---------------------------------------------------------------------------
// expmap0 over rows: out = project(tanh(||u||)/||u|| * u); writes bf16 and/or
// fp32 copies plus the projected row norm (needed by mobius_matvec).
// One block per row, blockDim.x == D (128 or 256).
// ---------------------------------------------------------------------------
__global__ void expmap_rows_kernel(const float* __restrict__ in,
                                   unsigned short* __restrict__ obf,
                                   float* __restrict__ of32,
                                   float* __restrict__ xn, int D) {
    __shared__ float sm[256];
    const int row = blockIdx.x, t = threadIdx.x;
    const size_t idx = (size_t)row * D + t;
    float u  = in[idx];
    float un = fmaxf(sqrtf(block_sum(u * u, sm)), TINY);
    float tn = tanhf(un);                       // ||tan_k(un)/un * u|| = tanh(un)
    float s  = tn / un;
    float rn = tn;
    if (rn > MAXNORM) { s *= MAXNORM / rn; rn = MAXNORM; }
    float val = s * u;
    if (obf)  obf[idx]  = f2bf(val);
    if (of32) of32[idx] = val;
    if (xn && t == 0) xn[row] = fmaxf(rn, TINY);
}

// ---------------------------------------------------------------------------
// WMMA bf16 GEMM:  out[N, Ncols] = Xb[N, KD] * Wb[Ncols, KD]^T  (f32 acc)
// One wave computes a 16x64 output strip (4 accumulators); KD is a compile-
// time constant (256) so the K-loop fully unrolls into 8 steps x 4 WMMAs.
// Software pipelined: step k+1 fragments are fetched before step k's WMMAs
// so the matrix pipe overlaps a full K-step of load latency.
// Fragment lane layouts used here (contiguous k, same order for A and B):
//   A (16x32 bf16): lane m=l&15; VGPRs 0-7 = 16 consecutive bf16 @ k0+hi*16
//                   -> 2x b128
//   B (32x16 bf16): lane n=l&15; VGPRs 0-7 = 16 consecutive bf16 @ k0+hi*16
//                   -> 2x b128
// ---------------------------------------------------------------------------
template <int KD>
__global__ void wmma_gemm_bf16_kernel(const unsigned short* __restrict__ Xb,
                                      const unsigned short* __restrict__ Wb,
                                      float* __restrict__ out,
                                      int Nrows, int Ncols) {
    const int lane = threadIdx.x;               // 0..31 (wave32)
    const int hi   = lane >> 4;                 // half-wave select
    const int ml   = lane & 15;
    const int n0   = (blockIdx.y * blockDim.y + threadIdx.y) * 64;
    if (n0 >= Ncols) return;
    const int m0 = blockIdx.x * 16;

    int mr = m0 + ml; if (mr >= Nrows) mr = Nrows - 1;      // clamp partial tile
    const uint4* Arow  = (const uint4*)(Xb + (size_t)mr * KD);  // 8 bf16 / granule
    const uint4* Brow0 = (const uint4*)(Wb + (size_t)(n0 +  0 + ml) * KD);
    const uint4* Brow1 = (const uint4*)(Wb + (size_t)(n0 + 16 + ml) * KD);
    const uint4* Brow2 = (const uint4*)(Wb + (size_t)(n0 + 32 + ml) * KD);
    const uint4* Brow3 = (const uint4*)(Wb + (size_t)(n0 + 48 + ml) * KD);

    auto loadA = [&](int k0) -> v8u {
        const uint4 lo = Arow[(k0 + (hi << 4)) >> 3];
        const uint4 hv = Arow[((k0 + (hi << 4)) >> 3) + 1];
        return (v8u){ lo.x, lo.y, lo.z, lo.w, hv.x, hv.y, hv.z, hv.w };
    };
    auto loadB = [&](const uint4* br, int k0) -> v8u {
        const int g = (k0 + (hi << 4)) >> 3;
        const uint4 lo = br[g];
        const uint4 hv = br[g + 1];
        return (v8u){ lo.x, lo.y, lo.z, lo.w, hv.x, hv.y, hv.z, hv.w };
    };

    constexpr int NSTEP = KD / 32;
    v8f acc[4] = {};

    v8u av  = loadA(0);
    v8u bv0 = loadB(Brow0, 0), bv1 = loadB(Brow1, 0);
    v8u bv2 = loadB(Brow2, 0), bv3 = loadB(Brow3, 0);

#pragma unroll
    for (int step = 0; step < NSTEP; ++step) {
        v8u av_n, bv0_n, bv1_n, bv2_n, bv3_n;
        const bool more = (step + 1 < NSTEP);
        if (more) {                              // prefetch next K-step
            const int kn = (step + 1) * 32;
            av_n  = loadA(kn);
            bv0_n = loadB(Brow0, kn);
            bv1_n = loadB(Brow1, kn);
            bv2_n = loadB(Brow2, kn);
            bv3_n = loadB(Brow3, kn);
        }
        const v16bf a = __builtin_bit_cast(v16bf, av);
        acc[0] = __builtin_amdgcn_wmma_f32_16x16x32_bf16(
                     false, a, false, __builtin_bit_cast(v16bf, bv0),
                     (short)0, acc[0], false, false);
        acc[1] = __builtin_amdgcn_wmma_f32_16x16x32_bf16(
                     false, a, false, __builtin_bit_cast(v16bf, bv1),
                     (short)0, acc[1], false, false);
        acc[2] = __builtin_amdgcn_wmma_f32_16x16x32_bf16(
                     false, a, false, __builtin_bit_cast(v16bf, bv2),
                     (short)0, acc[2], false, false);
        acc[3] = __builtin_amdgcn_wmma_f32_16x16x32_bf16(
                     false, a, false, __builtin_bit_cast(v16bf, bv3),
                     (short)0, acc[3], false, false);
        if (more) {
            av = av_n; bv0 = bv0_n; bv1 = bv1_n; bv2 = bv2_n; bv3 = bv3_n;
        }
    }

    // D layout: VGPR v -> row v + 8*hi, col = lane&15
    const bool full = (m0 + 16 <= Nrows);
#pragma unroll
    for (int t = 0; t < 4; ++t) {
        if (n0 + t * 16 >= Ncols) break;
        float* ot = out + (size_t)(m0 + (hi << 3)) * Ncols + (n0 + t * 16 + ml);
        if (full) {
#pragma unroll
            for (int v = 0; v < 8; ++v) ot[(size_t)v * Ncols] = acc[t][v];
        } else {
#pragma unroll
            for (int v = 0; v < 8; ++v)
                if (m0 + (hi << 3) + v < Nrows) ot[(size_t)v * Ncols] = acc[t][v];
        }
    }
}

// ---------------------------------------------------------------------------
// Fused: mobius_matvec norm-rescale + mobius_add(res, kb) + logmap0.
// In-place on the GEMM output (each thread owns its element).
// Mobius-add result norm computed analytically from x2, y2, xy.
// ---------------------------------------------------------------------------
__global__ void mobius_log_rows_kernel(const float* __restrict__ mv,
                                       const float* __restrict__ xn,
                                       const float* __restrict__ kb,
                                       float* __restrict__ xt, int D) {
    __shared__ float sm[256];
    const int row = blockIdx.x, t = threadIdx.x;
    const size_t idx = (size_t)row * D + t;

    // tan_k(||Mx||/||x|| * artan_k(||x||)) * Mx/||Mx||, then project
    float m   = mv[idx];
    float mvn = fmaxf(sqrtf(block_sum(m * m, sm)), TINY);
    float arg = mvn / xn[row] * artanh_c(xn[row]);
    float tn  = tanhf(arg);
    float s   = tn / mvn;
    float rn  = tn;
    if (rn > MAXNORM) { s *= MAXNORM / rn; rn = MAXNORM; }
    float rx = s * m;                           // res element

    // mobius_add(res, kb) with K = -1:
    // num = (1 + 2xy + y2)*x + (1 - x2)*y ; den = 1 + 2xy + x2*y2
    float y  = kb[t];
    float xy = block_sum(rx * y, sm);
    float y2 = block_sum(y * y, sm);
    float x2 = rn * rn;
    float den = fmaxf(1.0f + 2.0f * xy + x2 * y2, TINY);
    float c1  = (1.0f + 2.0f * xy + y2) / den;
    float c2  = (1.0f - x2) / den;
    float z   = c1 * rx + c2 * y;
    float zn2 = c1 * c1 * x2 + 2.0f * c1 * c2 * xy + c2 * c2 * y2;
    float zn  = fmaxf(sqrtf(zn2), TINY);
    if (zn > MAXNORM) { z *= MAXNORM / zn; zn = MAXNORM; }

    // logmap0
    xt[idx] = artanh_c(zn) / zn * z;
}

// ---------------------------------------------------------------------------
// Aggregation: agg[i] = dis[i]^2 * xt[i]  (self-loop term; also serves as the
// zero-init of the poisoned workspace), then atomic scatter over real edges.
// ---------------------------------------------------------------------------
__global__ void agg_init_kernel(const float* __restrict__ xt, const float* __restrict__ dis,
                                float* __restrict__ agg, int N, int D) {
    long long total = (long long)N * D;
    for (long long i = (long long)blockIdx.x * blockDim.x + threadIdx.x; i < total;
         i += (long long)gridDim.x * blockDim.x) {
        int r = (int)(i / D);
        float d = dis[r];
        agg[i] = d * d * xt[i];
    }
}

__global__ void agg_scatter_kernel(const int* __restrict__ row, const int* __restrict__ col,
                                   const float* __restrict__ xt, const float* __restrict__ dis,
                                   float* __restrict__ agg, int E, int D) {
    const int q = D >> 2;                       // float4 chunks per row
    long long total = (long long)E * q;
    for (long long i = (long long)blockIdx.x * blockDim.x + threadIdx.x; i < total;
         i += (long long)gridDim.x * blockDim.x) {
        int e = (int)(i / q);
        int f = ((int)(i % q)) << 2;
        int r = row[e], c = col[e];
        float w = dis[r] * dis[c];
        const float4 v = *(const float4*)(xt + (size_t)r * D + f);
        float* a = agg + (size_t)c * D + f;
        atomicAdd(a + 0, w * v.x);
        atomicAdd(a + 1, w * v.y);
        atomicAdd(a + 2, w * v.z);
        atomicAdd(a + 3, w * v.w);
    }
}

// ---------------------------------------------------------------------------
// host orchestration
// ---------------------------------------------------------------------------
extern "C" void kernel_launch(void* const* d_in, const int* in_sizes, int n_in,
                              void* d_out, int out_size, void* d_ws, size_t ws_size,
                              hipStream_t stream) {
    const float* x  = (const float*)d_in[0];
    const float* W1 = (const float*)d_in[1];
    const float* b1 = (const float*)d_in[2];
    const float* W2 = (const float*)d_in[3];
    const float* b2 = (const float*)d_in[4];
    const int*   ei = (const int*)d_in[5];

    const int HID = in_sizes[2];            // 256
    const int IN  = in_sizes[1] / HID;      // 256
    const int OUT = in_sizes[4];            // 128
    const int N   = in_sizes[0] / IN;       // 50000
    const int E   = in_sizes[5] / 2;        // 800000

    // workspace carve-up (~129 MB total)
    char* ws = (char*)d_ws;
    size_t off = 0;
    auto take = [&](size_t bytes) -> char* {
        char* p = ws + off;
        off = (off + bytes + 255) & ~(size_t)255;
        return p;
    };
    float*          dis = (float*)take((size_t)N * 4);
    float*          xn  = (float*)take((size_t)N * 4);
    float*          kb  = (float*)take(256 * 4);
    float*          kbn = (float*)take(256);
    unsigned short* Wb  = (unsigned short*)take((size_t)HID * IN * 2);
    unsigned short* Xb  = (unsigned short*)take((size_t)N * IN * 2);
    float*          mv  = (float*)take((size_t)N * HID * 4);   // reused in-place as xt
    float*          agg = (float*)take((size_t)N * HID * 4);

    const int* row = ei;
    const int* col = ei + E;

    // symmetric-normalization coefficients (shared by both layers)
    fill1_kernel<<<(N + 255) / 256, 256, 0, stream>>>(dis, N);
    deg_edges_kernel<<<(E + 255) / 256, 256, 0, stream>>>(col, dis, E);
    rsqrt_kernel<<<(N + 255) / 256, 256, 0, stream>>>(dis, N);

    // encoder input: x0 = expmap0(x) -> bf16 + row norms
    expmap_rows_kernel<<<N, IN, 0, stream>>>(x, Xb, nullptr, xn, IN);

    struct Layer { const float* W; const float* b; int Din; int Dout; bool last; };
    Layer layers[2] = { { W1, b1, IN,  HID, false },
                        { W2, b2, HID, OUT, true  } };

    for (int li = 0; li < 2; ++li) {
        const Layer& L = layers[li];

        // weights -> bf16, bias -> hyperbolic bias point kb = expmap0(b)
        int wn = L.Dout * L.Din;
        to_bf16_kernel<<<(wn + 255) / 256, 256, 0, stream>>>(L.W, Wb, wn);
        expmap_rows_kernel<<<1, L.Dout, 0, stream>>>(L.b, nullptr, kb, kbn, L.Dout);

        // mv = X * W^T on the bf16 WMMA pipe (f32 accumulate), KD = 256
        int wgroups = L.Dout / 64;                  // 16x64 strips across columns
        dim3 blk(32, wgroups < 4 ? wgroups : 4);
        dim3 grd((N + 15) / 16, (wgroups + blk.y - 1) / blk.y);
        wmma_gemm_bf16_kernel<256><<<grd, blk, 0, stream>>>(Xb, Wb, mv, N, L.Dout);

        // hyperbolic matvec rescale + bias add + logmap0 (in-place -> xt)
        mobius_log_rows_kernel<<<N, L.Dout, 0, stream>>>(mv, xn, kb, mv, L.Dout);

        // D^-1/2 A D^-1/2 aggregation in tangent space
        agg_init_kernel<<<4096, 256, 0, stream>>>(mv, dis, agg, N, L.Dout);
        agg_scatter_kernel<<<16384, 256, 0, stream>>>(row, col, mv, dis, agg, E, L.Dout);

        // expmap0 back to the ball
        if (!L.last) {
            expmap_rows_kernel<<<N, L.Dout, 0, stream>>>(agg, Xb, nullptr, xn, L.Dout);
        } else {
            expmap_rows_kernel<<<N, L.Dout, 0, stream>>>(agg, nullptr, (float*)d_out,
                                                         nullptr, L.Dout);
        }
    }
}
